// SoftExpertLayer_83116207112502
// MI455X (gfx1250) — matrix operand order repaired and checked by reading, not compiled, not run on hardware
//
#include <hip/hip_runtime.h>
#include <hip/hip_bf16.h>

#define BATCH 4
#define SEQ   2048
#define BS    8192            // BATCH*SEQ tokens
#define HID   1024
#define NEXP  16
#define SLOTS 128
#define ES    2048            // NEXP*SLOTS
#define FFN   2730
#define HPAD  2752            // FFN padded to multiple of 32 (async-copyable rows)
#define BK    32
#define LDT   40              // LDS row stride (bf16) for 32-wide K tiles

typedef __bf16 bf16;
typedef __attribute__((ext_vector_type(16))) __bf16 v16bf;
typedef __attribute__((ext_vector_type(8)))  __bf16 v8bf;
typedef __attribute__((ext_vector_type(8)))  float  v8f;

// ---------------- CDNA5 async global->LDS copy ----------------
__device__ __forceinline__ unsigned lds_off_of(const void* p) {
    // generic LDS address: low 32 bits are the LDS byte offset (aperture in high bits)
    return (unsigned)(unsigned long long)p;
}
__device__ __forceinline__ void async_copy16(bf16* dst_lds, const bf16* src_g) {
    unsigned l = lds_off_of(dst_lds);
    asm volatile("global_load_async_to_lds_b128 %0, %1, off"
                 :: "v"(l), "v"(src_g) : "memory");
}
__device__ __forceinline__ void wait_async0() {
    asm volatile("s_wait_asynccnt 0x0" ::: "memory");
}
// A tile (128 x 32 bf16) straight copy from row-major global, fully in-bounds
__device__ __forceinline__ void asyncA_rm(bf16* As, const bf16* g, int ld, int k0, int tid) {
#pragma unroll
    for (int it = 0; it < 2; ++it) {
        int idx = it * 256 + tid;
        int row = idx >> 2, ch = (idx & 3) << 3;
        async_copy16(As + row * LDT + ch, g + (size_t)row * ld + k0 + ch);
    }
}

// ---------------- WMMA fragment loads from LDS ----------------
__device__ __forceinline__ void frag_loadA(v16bf& a, const bf16* As, int m_sub, int lane) {
    int r  = lane & 15;
    int kh = (lane >> 4) << 3;
    const bf16* p = As + (m_sub + r) * LDT;
    union { v16bf v; v8bf h[2]; } u;
    u.h[0] = *(const v8bf*)(p + kh);
    u.h[1] = *(const v8bf*)(p + 16 + kh);
    a = u.v;
}
__device__ __forceinline__ void frag_loadB(v16bf& b, const bf16* Bs, int n_sub, int lane) {
    int c  = lane & 15;
    int kh = (lane >> 4) << 3;
    const bf16* p = Bs + (n_sub + c) * LDT + kh;
    union { v16bf v; v8bf h[2]; } u;
    u.h[0] = *(const v8bf*)(p);
    u.h[1] = *(const v8bf*)(p + 16);
    b = u.v;
}
__device__ __forceinline__ void gemm_step(const bf16* As, const bf16* Bs,
                                          v8f acc[2][4], int wm, int wn, int lane) {
    v16bf af[2], bfr[4];
#pragma unroll
    for (int i = 0; i < 2; ++i) frag_loadA(af[i], As, wm * 32 + i * 16, lane);
#pragma unroll
    for (int j = 0; j < 4; ++j) frag_loadB(bfr[j], Bs, wn * 64 + j * 16, lane);
#pragma unroll
    for (int i = 0; i < 2; ++i)
#pragma unroll
        for (int j = 0; j < 4; ++j)
            acc[i][j] = __builtin_amdgcn_wmma_f32_16x16x32_bf16(
                false, af[i], false, bfr[j], (short)0, acc[i][j], false, false);
}

// ---------------- VGPR-staged tile fetch/store (transpose / convert paths) ----------------
// bf16 k-major global [K][ldn] -> n-major LDS tile 128n x 32k (used for A of xtilda and B tiles)
struct StageBf { v8bf r[2]; };
__device__ __forceinline__ void fetch_tr_bf16(StageBf& st, const bf16* g, int ldn, int k0, int n0, int tid) {
#pragma unroll
    for (int it = 0; it < 2; ++it) {
        int idx = it * 256 + tid;
        int k = idx >> 4, seg = (idx & 15) << 3;
        st.r[it] = *(const v8bf*)(g + (size_t)(k0 + k) * ldn + n0 + seg);
    }
}
__device__ __forceinline__ void store_tr_bf16(const StageBf& st, bf16* T, int tid) {
#pragma unroll
    for (int it = 0; it < 2; ++it) {
        int idx = it * 256 + tid;
        int k = idx >> 4, seg = (idx & 15) << 3;
#pragma unroll
        for (int j = 0; j < 8; ++j) T[(seg + j) * LDT + k] = st.r[it][j];
    }
}
// fp32 weights [K][ldn] with 8B-aligned rows (w_up): guarded float2 fetch, no K tail
struct StageF2 { float2 r[8]; };
__device__ __forceinline__ void fetch_f32v2(StageF2& st, const float* g, int ldn, int k0, int n0,
                                            int Ndim, int tid) {
#pragma unroll
    for (int it = 0; it < 8; ++it) {
        int idx = it * 256 + tid;      // 0..2047
        int k = idx >> 6;              // 0..31
        int n = n0 + ((idx & 63) << 1);
        float2 v = make_float2(0.0f, 0.0f);
        if (n < Ndim) v = *(const float2*)(g + (size_t)(k0 + k) * ldn + n);
        st.r[it] = v;
    }
}
__device__ __forceinline__ void store_f32v2(const StageF2& st, bf16* Bs, int tid) {
#pragma unroll
    for (int it = 0; it < 8; ++it) {
        int idx = it * 256 + tid;
        int k = idx >> 6;
        int f2 = (idx & 63) << 1;
        Bs[(f2 + 0) * LDT + k] = (bf16)st.r[it].x;
        Bs[(f2 + 1) * LDT + k] = (bf16)st.r[it].y;
    }
}
// fp32 weights [K][1024] 16B-aligned rows (w_down): float4 fetch, K-tail guard only
struct StageF4 { float4 r[4]; };
__device__ __forceinline__ void fetch_f32v4(StageF4& st, const float* g, int ldn, int k0, int n0,
                                            int Kdim, int tid) {
#pragma unroll
    for (int it = 0; it < 4; ++it) {
        int idx = it * 256 + tid;      // 0..1023
        int k = idx >> 5;              // 0..31
        int f4 = (idx & 31) << 2;
        float4 v = make_float4(0.0f, 0.0f, 0.0f, 0.0f);
        if (k0 + k < Kdim) v = *(const float4*)(g + (size_t)(k0 + k) * ldn + n0 + f4);
        st.r[it] = v;
    }
}
__device__ __forceinline__ void store_f32v4(const StageF4& st, bf16* Bs, int tid) {
#pragma unroll
    for (int it = 0; it < 4; ++it) {
        int idx = it * 256 + tid;
        int k = idx >> 5;
        int f4 = (idx & 31) << 2;
        Bs[(f4 + 0) * LDT + k] = (bf16)st.r[it].x;
        Bs[(f4 + 1) * LDT + k] = (bf16)st.r[it].y;
        Bs[(f4 + 2) * LDT + k] = (bf16)st.r[it].z;
        Bs[(f4 + 3) * LDT + k] = (bf16)st.r[it].w;
    }
}

// ---------------- Softmax stage ----------------
__global__ __launch_bounds__(256) void k_disp_stats_partial(const float* __restrict__ logits,
                                                            float* __restrict__ pmax,
                                                            float* __restrict__ psum) {
    int es = blockIdx.x * 256 + threadIdx.x;
    int chunk = blockIdx.y;
    int b = blockIdx.z;
    const float* base = logits + (size_t)(b * SEQ + chunk * 128) * ES + es;
    float m = -3.4e38f, s = 0.0f;
    for (int t = 0; t < 128; ++t) {
        float v  = base[(size_t)t * ES];
        float nm = fmaxf(m, v);
        s = s * __expf(m - nm) + __expf(v - nm);
        m = nm;
    }
    int o = (b * 16 + chunk) * ES + es;
    pmax[o] = m; psum[o] = s;
}

__global__ __launch_bounds__(256) void k_disp_stats_reduce(const float* __restrict__ pmax,
                                                           const float* __restrict__ psum,
                                                           float* __restrict__ mx,
                                                           float* __restrict__ rs) {
    int gid = blockIdx.x * 256 + threadIdx.x;
    int b = gid >> 11, es = gid & (ES - 1);
    float m = -3.4e38f;
#pragma unroll
    for (int c = 0; c < 16; ++c) m = fmaxf(m, pmax[(b * 16 + c) * ES + es]);
    float s = 0.0f;
#pragma unroll
    for (int c = 0; c < 16; ++c) s += psum[(b * 16 + c) * ES + es] * __expf(pmax[(b * 16 + c) * ES + es] - m);
    mx[gid] = m;
    rs[gid] = 1.0f / s;
}

__global__ __launch_bounds__(256) void k_token_softmax(const float* __restrict__ logits,
                                                       const float* __restrict__ x,
                                                       const float* __restrict__ mx,
                                                       const float* __restrict__ rs,
                                                       bf16* __restrict__ comb,
                                                       bf16* __restrict__ disp,
                                                       bf16* __restrict__ xb) {
    int token = blockIdx.x, tid = threadIdx.x;
    int b = token >> 11;
    const float* row = logits + (size_t)token * ES;
    __shared__ float red[256];
    float v[8];
    float m = -3.4e38f;
#pragma unroll
    for (int i = 0; i < 8; ++i) { v[i] = row[tid + i * 256]; m = fmaxf(m, v[i]); }
    red[tid] = m; __syncthreads();
    for (int s = 128; s > 0; s >>= 1) { if (tid < s) red[tid] = fmaxf(red[tid], red[tid + s]); __syncthreads(); }
    m = red[0]; __syncthreads();
    float sum = 0.0f;
#pragma unroll
    for (int i = 0; i < 8; ++i) { v[i] = __expf(v[i] - m); sum += v[i]; }
    red[tid] = sum; __syncthreads();
    for (int s = 128; s > 0; s >>= 1) { if (tid < s) red[tid] += red[tid + s]; __syncthreads(); }
    float rinv = 1.0f / red[0];
    const float* mxb = mx + (size_t)b * ES;
    const float* rsb = rs + (size_t)b * ES;
#pragma unroll
    for (int i = 0; i < 8; ++i) {
        int c = tid + i * 256;
        comb[(size_t)token * ES + c] = (bf16)(v[i] * rinv);
        disp[(size_t)token * ES + c] = (bf16)(__expf(row[c] - mxb[c]) * rsb[c]);
    }
#pragma unroll
    for (int i = 0; i < 4; ++i) {
        int c = tid + i * 256;
        xb[(size_t)token * HID + c] = (bf16)x[(size_t)token * HID + c];
    }
}

// ---------------- GEMM kernels (double-buffered, async where pure-copy) ----------------
// x_tilda[es][hid] = dispatch^T (2048x8192) @ xb (8192x1024); both operands VGPR-staged transposes
__global__ __launch_bounds__(256) void k_gemm_xtilda(const bf16* __restrict__ disp,
                                                     const bf16* __restrict__ xb,
                                                     bf16* __restrict__ xt) {
    __shared__ __align__(16) bf16 As[2][128 * LDT];
    __shared__ __align__(16) bf16 Bs[2][128 * LDT];
    int tid = threadIdx.x, lane = tid & 31, wave = tid >> 5;
    int wm = wave & 3, wn = wave >> 2;
    int m0 = blockIdx.x * 128, n0 = blockIdx.y * 128;
    v8f acc[2][4];
#pragma unroll
    for (int i = 0; i < 2; ++i)
#pragma unroll
        for (int j = 0; j < 4; ++j) acc[i][j] = (v8f)(0.0f);

    StageBf sa, sb;
    fetch_tr_bf16(sa, disp, ES, 0, m0, tid);
    fetch_tr_bf16(sb, xb, HID, 0, n0, tid);
    store_tr_bf16(sa, As[0], tid);
    store_tr_bf16(sb, Bs[0], tid);
    __syncthreads();
    int p = 0;
    for (int k0 = 0; k0 < BS; k0 += BK) {
        int kn = k0 + BK;
        if (kn < BS) {
            fetch_tr_bf16(sa, disp, ES, kn, m0, tid);
            fetch_tr_bf16(sb, xb, HID, kn, n0, tid);
        }
        gemm_step(As[p], Bs[p], acc, wm, wn, lane);
        if (kn < BS) {
            store_tr_bf16(sa, As[p ^ 1], tid);
            store_tr_bf16(sb, Bs[p ^ 1], tid);
        }
        __syncthreads();
        p ^= 1;
    }
    int rl = (lane >> 4) << 3, cl = lane & 15;
#pragma unroll
    for (int i = 0; i < 2; ++i)
#pragma unroll
        for (int j = 0; j < 4; ++j)
#pragma unroll
            for (int v = 0; v < 8; ++v)
                xt[(size_t)(m0 + wm * 32 + i * 16 + rl + v) * HID + n0 + wn * 64 + j * 16 + cl] =
                    (bf16)acc[i][j][v];
}

// h[e] = silu(x_tilda[e] @ w_up[e]); A tile via async copy, B via float2-staged convert
__global__ __launch_bounds__(256) void k_gemm_up(const bf16* __restrict__ xt,
                                                 const float* __restrict__ w_up,
                                                 bf16* __restrict__ hbuf) {
    __shared__ __align__(16) bf16 As[2][128 * LDT];
    __shared__ __align__(16) bf16 Bs[2][128 * LDT];
    int tid = threadIdx.x, lane = tid & 31, wave = tid >> 5;
    int wm = wave & 3, wn = wave >> 2;
    int e = blockIdx.x, n0 = blockIdx.y * 128;
    const bf16*  A = xt + (size_t)e * 128 * HID;
    const float* W = w_up + (size_t)e * HID * FFN;
    v8f acc[2][4];
#pragma unroll
    for (int i = 0; i < 2; ++i)
#pragma unroll
        for (int j = 0; j < 4; ++j) acc[i][j] = (v8f)(0.0f);

    StageF2 sb;
    asyncA_rm(As[0], A, HID, 0, tid);
    fetch_f32v2(sb, W, FFN, 0, n0, FFN, tid);
    store_f32v2(sb, Bs[0], tid);
    wait_async0();
    __syncthreads();
    int p = 0;
    for (int k0 = 0; k0 < HID; k0 += BK) {
        int kn = k0 + BK;
        if (kn < HID) {
            asyncA_rm(As[p ^ 1], A, HID, kn, tid);
            fetch_f32v2(sb, W, FFN, kn, n0, FFN, tid);
        }
        gemm_step(As[p], Bs[p], acc, wm, wn, lane);
        if (kn < HID) {
            store_f32v2(sb, Bs[p ^ 1], tid);
            wait_async0();
        }
        __syncthreads();
        p ^= 1;
    }
    int rl = (lane >> 4) << 3, cl = lane & 15;
#pragma unroll
    for (int i = 0; i < 2; ++i)
#pragma unroll
        for (int j = 0; j < 4; ++j)
#pragma unroll
            for (int v = 0; v < 8; ++v) {
                int col = n0 + wn * 64 + j * 16 + cl;
                size_t off = ((size_t)e * 128 + wm * 32 + i * 16 + rl + v) * HPAD + col;
                if (col < FFN) {
                    float xv = acc[i][j][v];
                    hbuf[off] = (bf16)(xv / (1.0f + __expf(-xv)));   // SiLU
                } else if (col < HPAD) {
                    hbuf[off] = (bf16)0.0f;                          // zero-pad for async K loop
                }
            }
}

// e_out[e] (128x1024) = h[e] (128xHPAD, zero-padded) @ w_down[e] (2730x1024)
__global__ __launch_bounds__(256) void k_gemm_down(const bf16* __restrict__ hbuf,
                                                   const float* __restrict__ w_down,
                                                   bf16* __restrict__ eo) {
    __shared__ __align__(16) bf16 As[2][128 * LDT];
    __shared__ __align__(16) bf16 Bs[2][128 * LDT];
    int tid = threadIdx.x, lane = tid & 31, wave = tid >> 5;
    int wm = wave & 3, wn = wave >> 2;
    int e = blockIdx.x, n0 = blockIdx.y * 128;
    const bf16*  A = hbuf + (size_t)e * 128 * HPAD;
    const float* W = w_down + (size_t)e * FFN * HID;
    v8f acc[2][4];
#pragma unroll
    for (int i = 0; i < 2; ++i)
#pragma unroll
        for (int j = 0; j < 4; ++j) acc[i][j] = (v8f)(0.0f);

    StageF4 sb;
    asyncA_rm(As[0], A, HPAD, 0, tid);
    fetch_f32v4(sb, W, HID, 0, n0, FFN, tid);
    store_f32v4(sb, Bs[0], tid);
    wait_async0();
    __syncthreads();
    int p = 0;
    for (int k0 = 0; k0 < HPAD; k0 += BK) {   // padded A rows are zero beyond FFN
        int kn = k0 + BK;
        if (kn < HPAD) {
            asyncA_rm(As[p ^ 1], A, HPAD, kn, tid);
            fetch_f32v4(sb, W, HID, kn, n0, FFN, tid);
        }
        gemm_step(As[p], Bs[p], acc, wm, wn, lane);
        if (kn < HPAD) {
            store_f32v4(sb, Bs[p ^ 1], tid);
            wait_async0();
        }
        __syncthreads();
        p ^= 1;
    }
    int rl = (lane >> 4) << 3, cl = lane & 15;
#pragma unroll
    for (int i = 0; i < 2; ++i)
#pragma unroll
        for (int j = 0; j < 4; ++j)
#pragma unroll
            for (int v = 0; v < 8; ++v)
                eo[((size_t)e * 128 + wm * 32 + i * 16 + rl + v) * HID + n0 + wn * 64 + j * 16 + cl] =
                    (bf16)acc[i][j][v];
}

// y (8192x1024 fp32) = combine (8192x2048) @ e_out (2048x1024); A async, B bf16-staged transpose
__global__ __launch_bounds__(256) void k_gemm_out(const bf16* __restrict__ comb,
                                                  const bf16* __restrict__ eo,
                                                  float* __restrict__ y) {
    __shared__ __align__(16) bf16 As[2][128 * LDT];
    __shared__ __align__(16) bf16 Bs[2][128 * LDT];
    int tid = threadIdx.x, lane = tid & 31, wave = tid >> 5;
    int wm = wave & 3, wn = wave >> 2;
    int m0 = blockIdx.x * 128, n0 = blockIdx.y * 128;
    const bf16* A = comb + (size_t)m0 * ES;
    v8f acc[2][4];
#pragma unroll
    for (int i = 0; i < 2; ++i)
#pragma unroll
        for (int j = 0; j < 4; ++j) acc[i][j] = (v8f)(0.0f);

    StageBf sb;
    asyncA_rm(As[0], A, ES, 0, tid);
    fetch_tr_bf16(sb, eo, HID, 0, n0, tid);
    store_tr_bf16(sb, Bs[0], tid);
    wait_async0();
    __syncthreads();
    int p = 0;
    for (int k0 = 0; k0 < ES; k0 += BK) {
        int kn = k0 + BK;
        if (kn < ES) {
            asyncA_rm(As[p ^ 1], A, ES, kn, tid);
            fetch_tr_bf16(sb, eo, HID, kn, n0, tid);
        }
        gemm_step(As[p], Bs[p], acc, wm, wn, lane);
        if (kn < ES) {
            store_tr_bf16(sb, Bs[p ^ 1], tid);
            wait_async0();
        }
        __syncthreads();
        p ^= 1;
    }
    int rl = (lane >> 4) << 3, cl = lane & 15;
#pragma unroll
    for (int i = 0; i < 2; ++i)
#pragma unroll
        for (int j = 0; j < 4; ++j)
#pragma unroll
            for (int v = 0; v < 8; ++v)
                y[(size_t)(m0 + wm * 32 + i * 16 + rl + v) * HID + n0 + wn * 64 + j * 16 + cl] =
                    acc[i][j][v];
}

// ---------------- launch ----------------
extern "C" void kernel_launch(void* const* d_in, const int* in_sizes, int n_in,
                              void* d_out, int out_size, void* d_ws, size_t ws_size,
                              hipStream_t stream) {
    const float* x      = (const float*)d_in[0];
    const float* logits = (const float*)d_in[1];
    const float* w_up   = (const float*)d_in[2];
    const float* w_down = (const float*)d_in[3];
    (void)in_sizes; (void)n_in; (void)out_size; (void)ws_size;

    char* wsp = (char*)d_ws;
    size_t off = 0;
    auto take = [&](size_t bytes) -> void* {
        void* p = wsp + off;
        off += (bytes + 255) & ~(size_t)255;
        return p;
    };
    bf16*  xb   = (bf16*) take((size_t)BS * HID * 2);
    bf16*  disp = (bf16*) take((size_t)BS * ES * 2);
    bf16*  comb = (bf16*) take((size_t)BS * ES * 2);
    bf16*  xt   = (bf16*) take((size_t)ES * HID * 2);
    bf16*  hbuf = (bf16*) take((size_t)NEXP * SLOTS * HPAD * 2);
    bf16*  eo   = (bf16*) take((size_t)ES * HID * 2);
    float* pmax = (float*)take((size_t)BATCH * 16 * ES * 4);
    float* psum = (float*)take((size_t)BATCH * 16 * ES * 4);
    float* mx   = (float*)take((size_t)BATCH * ES * 4);
    float* rs   = (float*)take((size_t)BATCH * ES * 4);

    k_disp_stats_partial<<<dim3(8, 16, 4), 256, 0, stream>>>(logits, pmax, psum);
    k_disp_stats_reduce<<<32, 256, 0, stream>>>(pmax, psum, mx, rs);
    k_token_softmax<<<BS, 256, 0, stream>>>(logits, x, mx, rs, comb, disp, xb);
    k_gemm_xtilda<<<dim3(ES / 128, HID / 128), 256, 0, stream>>>(disp, xb, xt);
    k_gemm_up<<<dim3(NEXP, (FFN + 127) / 128), 256, 0, stream>>>(xt, w_up, hbuf);
    k_gemm_down<<<dim3(NEXP, HID / 128), 256, 0, stream>>>(hbuf, w_down, eo);
    k_gemm_out<<<dim3(BS / 128, HID / 128), 256, 0, stream>>>(comb, eo, (float*)d_out);
}
